// FastCASSBlock_35622458753248
// MI455X (gfx1250) — matrix operand order repaired and checked by reading, not compiled, not run on hardware
//
#include <hip/hip_runtime.h>
#include <cstddef>
#include <cstdint>

// ---------------- problem constants ----------------
#define BB   16
#define HH   64
#define WW   64
#define DIM  256
#define LL   4096           // H*W
#define DINNER 512
#define HID  256
#define G3   768            // 3*HID
#define HSTR 264            // padded LDS row stride for h (bank-conflict free)
#define XPSTEP (BB * G3)    // 12288 floats per recurrence step

// ---------------- CDNA5 types ----------------
typedef __bf16 bf16;
typedef __attribute__((ext_vector_type(16))) bf16  v16bf;
typedef __attribute__((ext_vector_type(8)))  bf16  v8bf;
typedef __attribute__((ext_vector_type(8)))  float v8f;
typedef __attribute__((ext_vector_type(4)))  unsigned int u32x4;
typedef __attribute__((ext_vector_type(8)))  int i32x8;
typedef __attribute__((ext_vector_type(4)))  int i32x4;

union FragA { v16bf v; v8bf h[2]; };

__device__ __forceinline__ v8f wmma_bf16(v16bf a, v16bf b, v8f c) {
  return __builtin_amdgcn_wmma_f32_16x16x32_bf16(false, a, false, b, (short)0, c,
                                                 false, false);
}

__device__ __forceinline__ unsigned short f2bf(float f) {
  unsigned u = __builtin_bit_cast(unsigned, f);
  u += 0x7FFFu + ((u >> 16) & 1u);          // round-to-nearest-even
  return (unsigned short)(u >> 16);
}

__device__ __forceinline__ float wave_sum(float v) {
#pragma unroll
  for (int m = 16; m >= 1; m >>= 1) v += __shfl_xor(v, m, 32);
  return v;
}

// fast sigmoid/tanh built on v_exp_f32 + v_rcp_f32 (short TRANS chains)
__device__ __forceinline__ float fast_sig(float x) {
  return __builtin_amdgcn_rcpf(1.f + __expf(-x));
}
__device__ __forceinline__ float fast_tanh(float x) {
  return 1.f - 2.f * __builtin_amdgcn_rcpf(1.f + __expf(2.f * x));
}

// LDS byte offset of a generic pointer into shared memory
__device__ __forceinline__ unsigned lds_off(const void* p) {
  return (unsigned)(unsigned long long)(__attribute__((address_space(3))) const char*)p;
}

// ---- Tensor Data Mover: 1-D tile (12288 x f32) global -> LDS ----
__device__ __forceinline__ void tdm_load_step(const float* gsrc, unsigned lds_byte) {
  unsigned long long ga = (unsigned long long)(uintptr_t)gsrc;
  u32x4 g0;
  g0[0] = 1u;                                        // count=1 (valid user D#)
  g0[1] = lds_byte;                                  // lds_addr
  g0[2] = (unsigned)(ga & 0xFFFFFFFFull);            // global_addr[31:0]
  g0[3] = (unsigned)((ga >> 32) & 0x1FFFFFFull)      // global_addr[56:32]
          | (2u << 30);                              // type=2 ("image")
  i32x8 g1;
  g1[0] = 0x00020000;                                // data_size=4B, no multicast
  g1[1] = (int)(((unsigned)XPSTEP & 0xFFFFu) << 16); // tensor_dim0 lo16
  g1[2] = (int)(1u << 16);                           // tensor_dim0 hi=0, tensor_dim1=1 lo
  g1[3] = (int)(((unsigned)XPSTEP & 0xFFFFu) << 16); // tile_dim0 = 12288
  g1[4] = 1;                                         // tile_dim1=1, tile_dim2=0
  g1[5] = XPSTEP;                                    // tensor_dim0_stride lo32
  g1[6] = 0;
  g1[7] = 0;
  i32x4 gz = {0, 0, 0, 0};
#if __has_include(<hip/amd_detail/amd_gfx1250_TDM.h>)
  i32x8 gz8 = {0, 0, 0, 0, 0, 0, 0, 0};
  __builtin_amdgcn_tensor_load_to_lds(g0, g1, gz, gz, gz8, 0);
#else
  __builtin_amdgcn_tensor_load_to_lds(g0, g1, gz, gz, 0);
#endif
}

// ---------------- workspace layout (bytes) ----------------
#define OFF_XN    ((size_t)0)                       // f32  [B,L,C]
#define OFF_GRAY  ((size_t)67108864)                // f32  [B,L]
#define OFF_IDX   ((size_t)67371008)                // i32  [B]
#define OFF_SEQ   ((size_t)67371264)                // bf16 [B*L,256]
#define OFF_Z     ((size_t)100925696)               // bf16 [B*L,512]
#define OFF_XPF   ((size_t)168034560)               // f32  t-major [L,B,768]
#define OFF_XPB   ((size_t)369361152)               // f32  t-major [L,B,768]
#define OFF_Y     ((size_t)570687744)               // bf16 [B*L,512]
#define OFF_WBF   ((size_t)637796608)               // bf16 weights pool
#define OFF_BIAS  ((size_t)640680192)               // f32 combined gate biases 2x768
#define WO_INW    ((size_t)0)
#define WO_WIHF   (WO_INW  + 512*256)
#define WO_WIHB   (WO_WIHF + 768*512)
#define WO_WHHF   (WO_WIHB + 768*512)
#define WO_WHHB   (WO_WHHF + 768*256)
#define WO_OUTW   (WO_WHHB + 768*256)

// =====================================================================
// fp32 -> bf16 conversion (weights)
// =====================================================================
__global__ void cvt_bf16_kernel(const float* __restrict__ s,
                                unsigned short* __restrict__ d, int n) {
  int i = blockIdx.x * 256 + threadIdx.x;
  if (i < n) d[i] = f2bf(s[i]);
}

// =====================================================================
// combined gate bias: comb[i] = bih[i] + (i<512 ? bhh[i] : 0)
// (r,z biases commute into the input side; n-gate bias is gated by r)
// =====================================================================
__global__ void bias_comb_kernel(const float* __restrict__ bih_f,
                                 const float* __restrict__ bhh_f,
                                 const float* __restrict__ bih_b,
                                 const float* __restrict__ bhh_b,
                                 float* __restrict__ comb) {
  int i = blockIdx.x * 256 + threadIdx.x;   // [0, 1536)
  if (i >= 2 * G3) return;
  int d = i / G3, k = i % G3;
  const float* bih = d ? bih_b : bih_f;
  const float* bhh = d ? bhh_b : bhh_f;
  comb[i] = bih[k] + (k < 512 ? bhh[k] : 0.f);
}

// =====================================================================
// LayerNorm over channels + gray = mean(xn) ; one wave per pixel
// =====================================================================
__global__ void ln_kernel(const float* __restrict__ x,
                          const float* __restrict__ gamma,
                          const float* __restrict__ beta,
                          float* __restrict__ xn, float* __restrict__ gray) {
  int wave = threadIdx.x >> 5;
  int lane = threadIdx.x & 31;
  size_t pix = (size_t)blockIdx.x * 8 + wave;
  const float* px = x + pix * DIM;
  float vals[8];
  float s = 0.f, s2 = 0.f;
#pragma unroll
  for (int k = 0; k < 8; ++k) {
    float v = px[lane + 32 * k];
    vals[k] = v; s += v; s2 += v * v;
  }
  s = wave_sum(s); s2 = wave_sum(s2);
  float mu = s * (1.f / DIM);
  float var = s2 * (1.f / DIM) - mu * mu;
  float rstd = rsqrtf(var + 1e-5f);
  float gs = 0.f;
  float* po = xn + pix * DIM;
#pragma unroll
  for (int k = 0; k < 8; ++k) {
    int c = lane + 32 * k;
    float v = (vals[k] - mu) * rstd * gamma[c] + beta[c];
    po[c] = v; gs += v;
  }
  gs = wave_sum(gs);
  if (lane == 0) gray[pix] = gs * (1.f / DIM);
}

// =====================================================================
// Gradient-direction selector; one block per batch.
// =====================================================================
__device__ __forceinline__ int refl_(int v) {
  return v < 0 ? -v : (v > 63 ? 126 - v : v);
}

__global__ void score_kernel(const float* __restrict__ gray,
                             const float* __restrict__ w1, const float* __restrict__ b1,
                             const float* __restrict__ w2, const float* __restrict__ b2,
                             int* __restrict__ idx_out) {
  __shared__ __align__(16) float G[LL];
  __shared__ __align__(16) float red[256 * 4];
  int b = blockIdx.x, tid = threadIdx.x;
  for (int i = tid; i < LL; i += 256) G[i] = gray[(size_t)b * LL + i];
  __syncthreads();
  float s0 = 0, s1 = 0, s2 = 0, s3 = 0;
  const float SC = 66.f / 64.f;
  for (int p = tid; p < LL; p += 256) {
    int i = p >> 6, j = p & 63;
    float src = fminf(fmaxf((i + 0.5f) * SC - 0.5f, 0.f), 65.f);
    int r0 = (int)floorf(src), r1 = min(r0 + 1, 65);
    float wr = src - (float)r0;
    int jc1 = refl_(j + 1), jc0 = refl_(j - 1);
    int rr0 = refl_(r0 - 1), rr1 = refl_(r1 - 1);
    float gh0 = fabsf(G[rr0 * 64 + jc1] - G[rr0 * 64 + jc0]);
    float gh1 = fabsf(G[rr1 * 64 + jc1] - G[rr1 * 64 + jc0]);
    float gh = gh0 * (1.f - wr) + gh1 * wr;
    float srcc = fminf(fmaxf((j + 0.5f) * SC - 0.5f, 0.f), 65.f);
    int c0 = (int)floorf(srcc), c1 = min(c0 + 1, 65);
    float wc = srcc - (float)c0;
    int ir1 = refl_(i + 1), ir0 = refl_(i - 1);
    int cc0 = refl_(c0 - 1), cc1 = refl_(c1 - 1);
    float gv0 = fabsf(G[ir1 * 64 + cc0] - G[ir0 * 64 + cc0]);
    float gv1 = fabsf(G[ir1 * 64 + cc1] - G[ir0 * 64 + cc1]);
    float gv = gv0 * (1.f - wc) + gv1 * wc;
    float gd = (gh + gv) * 0.5f, ga = fabsf(gh - gv);
    float wgt = ((i == 0 || i == 63) ? 2.f : 3.f) *
                ((j == 0 || j == 63) ? 2.f : 3.f) * (1.f / (9.f * LL));
    s0 += gh * wgt; s1 += gv * wgt; s2 += gd * wgt; s3 += ga * wgt;
  }
  red[tid * 4 + 0] = s0; red[tid * 4 + 1] = s1;
  red[tid * 4 + 2] = s2; red[tid * 4 + 3] = s3;
  __syncthreads();
  for (int st = 128; st > 0; st >>= 1) {
    if (tid < st)
      for (int q = 0; q < 4; ++q) red[tid * 4 + q] += red[(tid + st) * 4 + q];
    __syncthreads();
  }
  if (tid == 0) {
    float sc[4] = {red[0], red[1], red[2], red[3]};
    float h1[32];
    for (int o = 0; o < 32; ++o) {
      float v = b1[o];
      for (int k = 0; k < 4; ++k) v += sc[k] * w1[o * 4 + k];
      h1[o] = fmaxf(v, 0.f);
    }
    float best = -1e30f; int bi = 0;
    for (int o = 0; o < 4; ++o) {
      float v = b2[o];
      for (int k = 0; k < 32; ++k) v += h1[k] * w2[o * 32 + k];
      if (v > best) { best = v; bi = o; }
    }
    idx_out[b] = bi;
  }
}

// =====================================================================
// direction routing: seq[b,l,c] = xn[b, route(l), c]  -> bf16
// =====================================================================
__global__ void route_kernel(const float* __restrict__ xn,
                             const int* __restrict__ idx,
                             unsigned short* __restrict__ seq) {
  size_t g = (size_t)blockIdx.x * 256 + threadIdx.x;
  int b = (int)(g >> 20);
  int rem = (int)(g & ((1u << 20) - 1));
  int l = rem >> 8, c = rem & 255;
  int lsrc = (idx[b] == 1) ? ((l & 63) * 64 + (l >> 6)) : l;
  seq[g] = f2bf(xn[((size_t)b << 20) + (size_t)lsrc * 256 + c]);
}

// =====================================================================
// Generic bf16 WMMA GEMM: C[M,N] = A[M,K] * W[N,K]^T + bias
//  per-wave 32x64 C tile (2 A-frags x 4 B-frags -> 8 WMMA per k-step)
//  mode 0: bf16 row-major   mode 1: f32 t-major [l,b,N]   mode 2: f32+resid
// =====================================================================
__global__ void gemm_bf16_kernel(const unsigned short* __restrict__ A,
                                 const unsigned short* __restrict__ W,
                                 const float* __restrict__ bias,
                                 int M, int K, int N, int mode,
                                 unsigned short* __restrict__ outb,
                                 float* __restrict__ outf,
                                 const float* __restrict__ resid) {
  const int NG = N >> 6;
  int gw = blockIdx.x * 8 + (threadIdx.x >> 5);
  int lane = threadIdx.x & 31;
  int mt = gw / NG, ng = gw % NG;         // mt indexes 32-row tiles
  if (mt >= (M >> 5)) return;
  const int r = lane & 15, hi = lane >> 4;
  const size_t abase0 = ((size_t)mt * 32 + r) * K;
  const size_t abase1 = abase0 + (size_t)16 * K;
  const v8f zv = {0.f, 0.f, 0.f, 0.f, 0.f, 0.f, 0.f, 0.f};
  v8f acc[2][4] = {{zv, zv, zv, zv}, {zv, zv, zv, zv}};
  const int KT = K >> 5;
  for (int kt = 0; kt < KT; ++kt) {
    FragA a0, a1;
    const unsigned short* ap0 = A + abase0 + kt * 32 + hi * 16;
    const unsigned short* ap1 = A + abase1 + kt * 32 + hi * 16;
    a0.h[0] = *(const v8bf*)ap0;
    a0.h[1] = *(const v8bf*)(ap0 + 8);
    a1.h[0] = *(const v8bf*)ap1;
    a1.h[1] = *(const v8bf*)(ap1 + 8);
#pragma unroll
    for (int t = 0; t < 4; ++t) {
      int n = (ng * 4 + t) * 16 + r;
      v16bf bfr = *(const v16bf*)(W + (size_t)n * K + kt * 32 + hi * 16);
      acc[0][t] = wmma_bf16(a0.v, bfr, acc[0][t]);
      acc[1][t] = wmma_bf16(a1.v, bfr, acc[1][t]);
    }
  }
#pragma unroll
  for (int sub = 0; sub < 2; ++sub) {
#pragma unroll
    for (int t = 0; t < 4; ++t) {
      int col = (ng * 4 + t) * 16 + r;
      float bv = bias[col];
#pragma unroll
      for (int i = 0; i < 8; ++i) {
        size_t row = (size_t)mt * 32 + sub * 16 + i + 8 * hi;
        float v = acc[sub][t][i] + bv;
        if (mode == 0) {
          outb[row * N + col] = f2bf(v);
        } else if (mode == 1) {
          int bb = (int)(row >> 12), l = (int)(row & 4095);
          outf[((size_t)l * BB + bb) * (size_t)N + col] = v;
        } else {
          outf[row * N + col] = v + resid[row * N + col];
        }
      }
    }
  }
}

// =====================================================================
// biGRU recurrence: grid=2 (dir), block=768 (24 waves).
// whh^T register-resident; h bf16 in LDS (padded, conflict-free b128);
// per-step xp DMA'd one step ahead via the Tensor Data Mover (double-
// buffered LDS); r/z biases pre-fused into xp; n-gate bhh in 1 register;
// fp32 carried state in regs; TRANS-op gates (v_exp + v_rcp).
// =====================================================================
__global__ void gru_kernel(const float* __restrict__ xpf,
                           const float* __restrict__ xpb,
                           const unsigned short* __restrict__ whhf_bf,
                           const unsigned short* __restrict__ whhb_bf,
                           const float* __restrict__ bhhf,
                           const float* __restrict__ bhhb,
                           unsigned short* __restrict__ ybf) {
  extern __shared__ __align__(32) char smem[];
  float* xp_lds = (float*)smem;                              // 2 * 12288 f32
  float* gh_s   = (float*)(smem + 2 * XPSTEP * 4);           // 12288 f32
  unsigned short* h_s = (unsigned short*)(smem + 3 * XPSTEP * 4); // 16*HSTR

  const int dir = blockIdx.x;
  const float* xp = dir ? xpb : xpf;
  const unsigned short* whh_bf = dir ? whhb_bf : whhf_bf;
  const float* bhh = dir ? bhhb : bhhf;
  const int tid = threadIdx.x;
  const int wave = tid >> 5, lane = tid & 31;
  const int r = lane & 15, hi = lane >> 4;
  const int colOff = dir * HID;

  for (int i = tid; i < BB * HSTR; i += 768) h_s[i] = 0;

  // static per-thread (b,j) map: j == tid&255 for every q (768 = 3*256)
  const int j = tid & 255;
  const float bhhn = bhh[512 + j];        // n-gate hidden bias, 1 register
  const int nq = (tid < 256) ? 6 : 5;
  int   oarr[6], hsI[6];
  size_t ybase[6];
  float hreg[6];
#pragma unroll
  for (int q = 0; q < 6; ++q) {
    int idx = tid + 768 * q;
    int bb = (idx >> 8) & 15;
    oarr[q] = bb * G3;
    hsI[q] = bb * HSTR + j;
    ybase[q] = ((size_t)bb * LL) * DINNER + colOff + j;
    hreg[q] = 0.f;
  }

  // register-resident whh^T fragments: 2 N-tiles x 8 K-tiles (128 VGPRs)
  v16bf Bf[2][8];
#pragma unroll
  for (int tl = 0; tl < 2; ++tl) {
    int n = (wave * 2 + tl) * 16 + r;
#pragma unroll
    for (int kt = 0; kt < 8; ++kt)
      Bf[tl][kt] = *(const v16bf*)(whh_bf + (size_t)n * HID + kt * 32 + hi * 16);
  }

  // prologue: DMA step-0 xp tile into buffer 0
  if (wave == 0) {
    int t0 = dir ? (LL - 1) : 0;
    tdm_load_step(xp + (size_t)t0 * XPSTEP, lds_off(xp_lds));
  }
  __syncthreads();

  const v8f zv = {0.f, 0.f, 0.f, 0.f, 0.f, 0.f, 0.f, 0.f};
  for (int s = 0; s < LL; ++s) {
    int t = dir ? (LL - 1 - s) : s;
    // ---- stage 1: gh = h @ whh^T  (M=16, N=768, K=256) ----
    v8f acc0 = zv, acc1 = zv;
#pragma unroll
    for (int kt = 0; kt < 8; ++kt) {
      FragA a;
      const unsigned short* hp = h_s + r * HSTR + kt * 32 + hi * 16;
      a.h[0] = *(const v8bf*)hp;
      a.h[1] = *(const v8bf*)(hp + 8);
      acc0 = wmma_bf16(a.v, Bf[0][kt], acc0);
      acc1 = wmma_bf16(a.v, Bf[1][kt], acc1);
    }
#pragma unroll
    for (int tl = 0; tl < 2; ++tl) {
      int n = (wave * 2 + tl) * 16 + r;
      v8f av = tl ? acc1 : acc0;
#pragma unroll
      for (int i = 0; i < 8; ++i) gh_s[(i + 8 * hi) * G3 + n] = av[i];
    }
    // ---- TDM: launch next step's xp tile, wait for current one ----
    if (wave == 0) {
      if (s + 1 < LL) {
        int tn = dir ? (t - 1) : (t + 1);
        tdm_load_step(xp + (size_t)tn * XPSTEP,
                      lds_off(xp_lds + ((s + 1) & 1) * XPSTEP));
        __builtin_amdgcn_s_wait_tensorcnt((short)1);  // current tile done
      } else {
        __builtin_amdgcn_s_wait_tensorcnt((short)0);
      }
    }
    __syncthreads();
    // ---- stage 2: gate math (LDS only + one y store) ----
    const float* xpt = xp_lds + (s & 1) * XPSTEP;
    for (int q = 0; q < nq; ++q) {
      int o = oarr[q];
      float xr = xpt[o + j], xz = xpt[o + 256 + j], xnv = xpt[o + 512 + j];
      float rg = fast_sig(xr + gh_s[o + j]);
      float zg = fast_sig(xz + gh_s[o + 256 + j]);
      float ng = fast_tanh(xnv + rg * (gh_s[o + 512 + j] + bhhn));
      float hnew = (1.f - zg) * ng + zg * hreg[q];
      hreg[q] = hnew;
      h_s[hsI[q]] = f2bf(hnew);
      ybf[ybase[q] + (size_t)t * DINNER] = f2bf(hnew);
    }
    __syncthreads();
  }
}

// =====================================================================
// host launcher
// =====================================================================
extern "C" void kernel_launch(void* const* d_in, const int* in_sizes, int n_in,
                              void* d_out, int out_size, void* d_ws, size_t ws_size,
                              hipStream_t stream) {
  (void)in_sizes; (void)n_in; (void)out_size; (void)ws_size;
  const float* x        = (const float*)d_in[0];
  const float* ln_gamma = (const float*)d_in[1];
  const float* ln_beta  = (const float*)d_in[2];
  const float* mlp_w1   = (const float*)d_in[3];
  const float* mlp_b1   = (const float*)d_in[4];
  const float* mlp_w2   = (const float*)d_in[5];
  const float* mlp_b2   = (const float*)d_in[6];
  const float* in_w     = (const float*)d_in[7];
  const float* in_b     = (const float*)d_in[8];
  const float* wih_f    = (const float*)d_in[9];
  const float* whh_f    = (const float*)d_in[10];
  const float* bih_f    = (const float*)d_in[11];
  const float* bhh_f    = (const float*)d_in[12];
  const float* wih_b    = (const float*)d_in[13];
  const float* whh_b    = (const float*)d_in[14];
  const float* bih_b    = (const float*)d_in[15];
  const float* bhh_b    = (const float*)d_in[16];
  const float* out_w    = (const float*)d_in[17];
  const float* out_b    = (const float*)d_in[18];
  float* out = (float*)d_out;

  char* ws = (char*)d_ws;
  float*          xn   = (float*)(ws + OFF_XN);
  float*          gray = (float*)(ws + OFF_GRAY);
  int*            idx  = (int*)(ws + OFF_IDX);
  unsigned short* seq  = (unsigned short*)(ws + OFF_SEQ);
  unsigned short* z    = (unsigned short*)(ws + OFF_Z);
  float*          xpf  = (float*)(ws + OFF_XPF);
  float*          xpb  = (float*)(ws + OFF_XPB);
  unsigned short* ybf  = (unsigned short*)(ws + OFF_Y);
  unsigned short* wpool = (unsigned short*)(ws + OFF_WBF);
  float*          comb = (float*)(ws + OFF_BIAS);   // [2][768]
  unsigned short* inw_bf  = wpool + WO_INW;
  unsigned short* wihf_bf = wpool + WO_WIHF;
  unsigned short* wihb_bf = wpool + WO_WIHB;
  unsigned short* whhf_bf = wpool + WO_WHHF;
  unsigned short* whhb_bf = wpool + WO_WHHB;
  unsigned short* outw_bf = wpool + WO_OUTW;

  cvt_bf16_kernel<<<(512*256 + 255)/256, 256, 0, stream>>>(in_w,  inw_bf,  512*256);
  cvt_bf16_kernel<<<(768*512 + 255)/256, 256, 0, stream>>>(wih_f, wihf_bf, 768*512);
  cvt_bf16_kernel<<<(768*512 + 255)/256, 256, 0, stream>>>(wih_b, wihb_bf, 768*512);
  cvt_bf16_kernel<<<(768*256 + 255)/256, 256, 0, stream>>>(whh_f, whhf_bf, 768*256);
  cvt_bf16_kernel<<<(768*256 + 255)/256, 256, 0, stream>>>(whh_b, whhb_bf, 768*256);
  cvt_bf16_kernel<<<(256*512 + 255)/256, 256, 0, stream>>>(out_w, outw_bf, 256*512);
  bias_comb_kernel<<<6, 256, 0, stream>>>(bih_f, bhh_f, bih_b, bhh_b, comb);

  ln_kernel<<<8192, 256, 0, stream>>>(x, ln_gamma, ln_beta, xn, gray);
  score_kernel<<<BB, 256, 0, stream>>>(gray, mlp_w1, mlp_b1, mlp_w2, mlp_b2, idx);
  route_kernel<<<65536, 256, 0, stream>>>(xn, idx, seq);

  const int M = BB * LL;  // 65536
  // in_proj (K=256, N=512) -> bf16 z
  gemm_bf16_kernel<<<(M/32)*(512/64)/8, 256, 0, stream>>>(
      seq, inw_bf, in_b, M, 256, 512, 0, z, nullptr, nullptr);
  // input-side gate GEMMs with fused r/z hidden biases (K=512, N=768)
  gemm_bf16_kernel<<<(M/32)*(768/64)/8, 256, 0, stream>>>(
      z, wihf_bf, comb, M, 512, 768, 1, nullptr, xpf, nullptr);
  gemm_bf16_kernel<<<(M/32)*(768/64)/8, 256, 0, stream>>>(
      z, wihb_bf, comb + G3, M, 512, 768, 1, nullptr, xpb, nullptr);

  // biGRU recurrence, both directions concurrently; dynamic LDS ~152KB
  const size_t gru_lds = (size_t)3 * XPSTEP * 4 + (size_t)BB * HSTR * 2;
  gru_kernel<<<2, 768, gru_lds, stream>>>(xpf, xpb, whhf_bf, whhb_bf,
                                          bhh_f, bhh_b, ybf);

  // out_proj + residual (K=512, N=256) -> d_out
  gemm_bf16_kernel<<<(M/32)*(256/64)/8, 256, 0, stream>>>(
      ybf, outw_bf, out_b, M, 512, 256, 2, nullptr, out, x);
}
